// CMPNN_Encoder_73151882985858
// MI455X (gfx1250) — compile-verified, matrix operand order checked by reading, not run on hardware
//
#include <hip/hip_runtime.h>

// ---------------------------------------------------------------------------
// CDNA5 (gfx1250) CMPNN encoder: all dense GEMMs on v_wmma_f32_16x16x32_bf16,
// segment sums via global_atomic_add_f32, fused GRU/sigmoid/scatter epilogues.
// ---------------------------------------------------------------------------

typedef __attribute__((ext_vector_type(16))) __bf16 bf16x16;
typedef __attribute__((ext_vector_type(8)))  __bf16 bf16x8;
typedef __attribute__((ext_vector_type(8)))  float  f32x8;

static constexpr int F_NODE = 133, F_BOND = 14, DD = 256;
static constexpr int NNODE = 100000, NBOND = 200000, NNB = 600000;
static constexpr int KP_NODE = 160;   // 133 -> pad 160 (5 k-tiles of 32)
static constexpr int KP_BOND = 160;   // 147 -> pad 160
static constexpr int KT160 = 5, KT256 = 8;

enum { EPI_HS = 0, EPI_HS_BF = 1, EPI_SIG = 2, EPI_GRU = 3, EPI_RF = 4 };

__device__ __forceinline__ float hswish(float x) {
  float t = x + 3.0f;
  t = fminf(fmaxf(t, 0.0f), 6.0f);
  return x * t * (1.0f / 6.0f);
}
__device__ __forceinline__ float sigm(float x) { return 1.0f / (1.0f + __expf(-x)); }

// ------------------------- weight packing ----------------------------------
// Pack W[k, n] (f32, row-major, 256 cols) into WMMA-B fragment order so each
// lane of a wave does one contiguous 32B load per (k-tile, n-tile) block.
// Block layout: blk = kt*16 + nt ; within blk: lane*16 + e (bf16 elements).
__global__ void pack_weight(const float* __restrict__ src, int rowOffset,
                            int kAct, int kPad, __bf16* __restrict__ dst) {
  int idx = blockIdx.x * blockDim.x + threadIdx.x;
  int total = kPad * DD;
  if (idx >= total) return;
  int blk = idx >> 9;            // /512
  int within = idx & 511;
  int lane = within >> 4;
  int e = within & 15;
  int kt = blk >> 4;             // 16 n-tiles per k-tile
  int nt = blk & 15;
  int half = lane >> 4;
  int n = nt * 16 + (lane & 15);
  int k = kt * 32 + (e & 7) + ((e >> 3) << 4) + half * 8;
  float v = (k < kAct) ? src[(size_t)(k + rowOffset) * DD + n] : 0.0f;
  dst[idx] = (__bf16)v;
}

// --------------------- bf16 operand builders -------------------------------
__global__ void build_node_bf(const float* __restrict__ node,
                              __bf16* __restrict__ dst) {
  int idx = blockIdx.x * blockDim.x + threadIdx.x;
  if (idx >= NNODE * KP_NODE) return;
  int r = idx / KP_NODE, c = idx % KP_NODE;
  float v = (c < F_NODE) ? node[(size_t)r * F_NODE + c] : 0.0f;
  dst[idx] = (__bf16)v;
}

__global__ void build_ib_bf(const float* __restrict__ node,
                            const float* __restrict__ bond,
                            const int* __restrict__ iIdx,
                            __bf16* __restrict__ dst) {
  int idx = blockIdx.x * blockDim.x + threadIdx.x;
  if (idx >= NBOND * KP_BOND) return;
  int r = idx / KP_BOND, c = idx % KP_BOND;
  float v = 0.0f;
  if (c < F_NODE)                 v = node[(size_t)iIdx[r] * F_NODE + c];
  else if (c < F_NODE + F_BOND)   v = bond[(size_t)r * F_BOND + (c - F_NODE)];
  dst[idx] = (__bf16)v;
}

__global__ void f32_to_bf16_k(const float* __restrict__ src,
                              __bf16* __restrict__ dst, int n4) {
  int i = blockIdx.x * blockDim.x + threadIdx.x;
  if (i >= n4) return;
  float4 v = ((const float4*)src)[i];
  int b = i * 4;
  dst[b + 0] = (__bf16)v.x; dst[b + 1] = (__bf16)v.y;
  dst[b + 2] = (__bf16)v.z; dst[b + 3] = (__bf16)v.w;
}

__global__ void zero_f32(float* __restrict__ p, int n4) {
  int i = blockIdx.x * blockDim.x + threadIdx.x;
  if (i < n4) ((float4*)p)[i] = make_float4(0.f, 0.f, 0.f, 0.f);
}

// ----------------------- segment sum (atomic) ------------------------------
// dst[dstIdx[e], :] += src[srcIdx?srcIdx[e]:e, :]   (256 cols, 4 cols/thread)
__global__ void seg_sum(const float* __restrict__ src,
                        const int* __restrict__ srcIdx,
                        const int* __restrict__ dstIdx,
                        float* __restrict__ dst, int nE) {
  int t = blockIdx.x * blockDim.x + threadIdx.x;
  int e = t >> 6;
  if (e >= nE) return;
  int c = (t & 63) << 2;
  int sr = srcIdx ? srcIdx[e] : e;
  float4 v = *(const float4*)(src + (size_t)sr * DD + c);
  float* d = dst + (size_t)dstIdx[e] * DD + c;
  atomicAdd(d + 0, v.x); atomicAdd(d + 1, v.y);
  atomicAdd(d + 2, v.z); atomicAdd(d + 3, v.w);
}

// --------------------------- WMMA GEMM -------------------------------------
struct GArgs {
  const __bf16* A0; const int* g0; const __bf16* B0; int lda0; int kt0;
  const __bf16* A1; const int* g1; const __bf16* B1; int lda1; int kt1;
  const __bf16* A2; const int* g2; const __bf16* B2; int lda2; int kt2;
  int M;
  const float* bias;
  float*  out;       // f32 output [M,256]
  __bf16* outBf;     // optional bf16 mirror
  const float* zBuf; // GRU: z
  const float* sBuf; // GRU: s
  const float* messF;          // R-fused: mess_bond f32
  const int* gIJ; const int* gKI;
  float* rDst;                 // R-fused: atomic destination
};

__device__ __forceinline__ void run_pair(f32x8 acc[8],
                                         const __bf16* __restrict__ A, int lda,
                                         int kTiles, const int* __restrict__ g,
                                         const __bf16* __restrict__ B,
                                         int rowClamped, int lane, int half,
                                         int wcol) {
  int grow = g ? g[rowClamped] : rowClamped;
  const __bf16* arow = A + (size_t)grow * lda;
  const __bf16* bcol = B + (size_t)wcol * 8 * 512 + (size_t)lane * 16;
  for (int kt = 0; kt < kTiles; ++kt) {
    // A 16x32 bf16 fragment: lane holds row (lane&15); K = {0..7,16..23}+half*8
    bf16x8 lo = *(const bf16x8*)(arow + kt * 32 + half * 8);
    bf16x8 hi = *(const bf16x8*)(arow + kt * 32 + 16 + half * 8);
    bf16x16 afrag;
#pragma unroll
    for (int i = 0; i < 8; ++i) { afrag[i] = lo[i]; afrag[i + 8] = hi[i]; }
    const __bf16* bb = bcol + (size_t)kt * 8192;  // 16 n-tiles * 512 per kt
#pragma unroll
    for (int c = 0; c < 8; ++c) {
      bf16x16 bfrag = *(const bf16x16*)(bb + (size_t)c * 512);
      acc[c] = __builtin_amdgcn_wmma_f32_16x16x32_bf16(
          false, afrag, false, bfrag, (short)0, acc[c], false, false);
    }
  }
}

template <int NP, int EPI>
__global__ __launch_bounds__(256) void wmma_gemm(GArgs a) {
  const int lane = threadIdx.x & 31;
  const int wave = threadIdx.x >> 5;
  const int wrow = wave & 3;   // 4 row-groups of 16 rows
  const int wcol = wave >> 2;  // 2 col-groups of 128 cols
  const int m0 = blockIdx.x * 64 + wrow * 16;
  const int half = lane >> 4;
  const int lmod = lane & 15;

  const int rowA = m0 + lmod;
  const int rowClamped = (rowA < a.M) ? rowA : 0;

  f32x8 acc[8] = {};

  run_pair(acc, a.A0, a.lda0, a.kt0, a.g0, a.B0, rowClamped, lane, half, wcol);
  if (NP >= 2)
    run_pair(acc, a.A1, a.lda1, a.kt1, a.g1, a.B1, rowClamped, lane, half, wcol);
  if (NP >= 3)
    run_pair(acc, a.A2, a.lda2, a.kt2, a.g2, a.B2, rowClamped, lane, half, wcol);

  const int colBase = wcol * 128;
#pragma unroll
  for (int c = 0; c < 8; ++c) {
#pragma unroll
    for (int v = 0; v < 8; ++v) {
      int row = m0 + v + half * 8;          // C/D layout: lanes 16-31 -> M+8
      int col = colBase + c * 16 + lmod;
      if (row < a.M) {
        float x = acc[c][v] + a.bias[col];
        size_t idx = (size_t)row * DD + col;
        if (EPI == EPI_HS) {
          a.out[idx] = hswish(x);
        } else if (EPI == EPI_HS_BF) {
          float o = hswish(x);
          a.out[idx] = o;
          a.outBf[idx] = (__bf16)o;
        } else if (EPI == EPI_SIG) {
          a.out[idx] = sigm(x);
        } else if (EPI == EPI_GRU) {
          float m = tanhf(x);
          float zv = a.zBuf[idx];
          float sv = a.sBuf[idx];
          float o = (1.0f - zv) * sv + zv * m;
          a.out[idx] = o;
          a.outBf[idx] = (__bf16)o;
        } else if (EPI == EPI_RF) {
          // r_ki = sigmoid(x); scatter-add r_ki * mess_ki into r[ij]
          float r = sigm(x);
          int ki = a.gKI[row];
          int ij = a.gIJ[row];
          float mv = a.messF[(size_t)ki * DD + col];
          atomicAdd(&a.rDst[(size_t)ij * DD + col], r * mv);
        }
      }
    }
  }
}

// --------------------------------------------------------------------------
extern "C" void kernel_launch(void* const* d_in, const int* in_sizes, int n_in,
                              void* d_out, int out_size, void* d_ws,
                              size_t ws_size, hipStream_t stream) {
  (void)in_sizes; (void)n_in; (void)out_size; (void)ws_size;

  const float* node = (const float*)d_in[0];
  const float* bond = (const float*)d_in[1];
  const int* connect = (const int*)d_in[2];
  const int* bn = (const int*)d_in[3];
  const float* w_node = (const float*)d_in[4];
  const float* b_node = (const float*)d_in[5];
  const float* w_node_final = (const float*)d_in[6];
  const float* b_node_final = (const float*)d_in[7];
  const float* w_bond = (const float*)d_in[8];
  const float* b_bond = (const float*)d_in[9];
  const float* w_bond_final = (const float*)d_in[10];
  const float* b_bond_final = (const float*)d_in[11];
  const float* w_z = (const float*)d_in[12];
  const float* b_z = (const float*)d_in[13];
  const float* w_r = (const float*)d_in[14];
  const float* b_r = (const float*)d_in[15];
  const float* u_w = (const float*)d_in[16];
  const float* w_m = (const float*)d_in[17];
  const float* b_m = (const float*)d_in[18];
  const float* w_n = (const float*)d_in[19];
  const float* b_n = (const float*)d_in[20];
  const float* u_n = (const float*)d_in[21];

  const int* i_idx = connect;                // connect[0]
  const int* j_idx = connect + NBOND;        // connect[1]
  const int* ij_idx = bn;                    // bond_neighbour[0]
  const int* ki_idx = bn + NNB;              // bond_neighbour[1]

  float* outNode = (float*)d_out;
  float* outBond = outNode + (size_t)NNODE * DD;

  // ----- workspace carve-up -----
  char* wp = (char*)d_ws;
  auto carve = [&](size_t bytes) -> char* {
    char* p = wp;
    wp += (bytes + 255) & ~size_t(255);
    return p;
  };
  __bf16* node_bf = (__bf16*)carve((size_t)NNODE * KP_NODE * 2);
  __bf16* ib_bf   = (__bf16*)carve((size_t)NBOND * KP_BOND * 2);
  float*  mb_f    = (float*)carve((size_t)NBOND * DD * 4);  // mess_bond f32
  __bf16* mb_bf   = (__bf16*)carve((size_t)NBOND * DD * 2);
  float*  s_f     = (float*)carve((size_t)NBOND * DD * 4);
  __bf16* s_bf    = (__bf16*)carve((size_t)NBOND * DD * 2);
  float*  r_f     = (float*)carve((size_t)NBOND * DD * 4);
  __bf16* r_bf    = (__bf16*)carve((size_t)NBOND * DD * 2);
  float*  z_f     = (float*)carve((size_t)NBOND * DD * 4);
  float*  mn_f    = (float*)carve((size_t)NNODE * DD * 4);  // mess_node f32
  __bf16* mn_bf   = (__bf16*)carve((size_t)NNODE * DD * 2);
  float*  ag_f    = (float*)carve((size_t)NNODE * DD * 4);  // aggr_node
  __bf16* ag_bf   = (__bf16*)carve((size_t)NNODE * DD * 2);
  auto packbuf = [&](int kpad) { return (__bf16*)carve((size_t)kpad * DD * 2); };
  __bf16* pk_wnode = packbuf(160);
  __bf16* pk_wbond = packbuf(160);
  __bf16* pk_wz_lo = packbuf(160); __bf16* pk_wz_hi = packbuf(256);
  __bf16* pk_wr_lo = packbuf(160); __bf16* pk_wr_hi = packbuf(256);
  __bf16* pk_wm    = packbuf(160);
  __bf16* pk_u     = packbuf(256);
  __bf16* pk_wn    = packbuf(160);
  __bf16* pk_un_lo = packbuf(256); __bf16* pk_un_hi = packbuf(256);
  __bf16* pk_wbf_lo = packbuf(160); __bf16* pk_wbf_hi = packbuf(256);
  __bf16* pk_wnf_lo = packbuf(160); __bf16* pk_wnf_hi = packbuf(256);

  // ----- pack weights (split concat-GEMMs into lo/hi pieces) -----
  auto packW = [&](const float* src, int ro, int kact, int kpad, __bf16* dst) {
    int total = kpad * DD;
    pack_weight<<<(total + 255) / 256, 256, 0, stream>>>(src, ro, kact, kpad, dst);
  };
  packW(w_node, 0, 133, 160, pk_wnode);
  packW(w_bond, 0, 147, 160, pk_wbond);
  packW(w_z, 0, 147, 160, pk_wz_lo);   packW(w_z, 147, 256, 256, pk_wz_hi);
  packW(w_r, 0, 147, 160, pk_wr_lo);   packW(w_r, 147, 256, 256, pk_wr_hi);
  packW(w_m, 0, 147, 160, pk_wm);
  packW(u_w, 0, 256, 256, pk_u);
  packW(w_n, 0, 133, 160, pk_wn);
  packW(u_n, 0, 256, 256, pk_un_lo);   packW(u_n, 256, 256, 256, pk_un_hi);
  packW(w_bond_final, 0, 147, 160, pk_wbf_lo);
  packW(w_bond_final, 147, 256, 256, pk_wbf_hi);
  packW(w_node_final, 0, 133, 160, pk_wnf_lo);
  packW(w_node_final, 133, 256, 256, pk_wnf_hi);

  // ----- bf16 operand build -----
  build_node_bf<<<(NNODE * KP_NODE + 255) / 256, 256, 0, stream>>>(node, node_bf);
  build_ib_bf<<<(NBOND * KP_BOND + 255) / 256, 256, 0, stream>>>(node, bond, i_idx, ib_bf);

  auto cvt = [&](const float* src, __bf16* dst, size_t n) {
    int n4 = (int)(n / 4);
    f32_to_bf16_k<<<(n4 + 255) / 256, 256, 0, stream>>>(src, dst, n4);
  };
  auto zero = [&](float* p, size_t n) {
    int n4 = (int)(n / 4);
    zero_f32<<<(n4 + 255) / 256, 256, 0, stream>>>(p, n4);
  };

  GArgs ga = {};
  auto setP0 = [&](const __bf16* A, int lda, int kt, const int* g, const __bf16* B) {
    ga.A0 = A; ga.lda0 = lda; ga.kt0 = kt; ga.g0 = g; ga.B0 = B; };
  auto setP1 = [&](const __bf16* A, int lda, int kt, const int* g, const __bf16* B) {
    ga.A1 = A; ga.lda1 = lda; ga.kt1 = kt; ga.g1 = g; ga.B1 = B; };
  auto setP2 = [&](const __bf16* A, int lda, int kt, const int* g, const __bf16* B) {
    ga.A2 = A; ga.lda2 = lda; ga.kt2 = kt; ga.g2 = g; ga.B2 = B; };

  const dim3 gE((NBOND + 63) / 64), gN((NNODE + 63) / 64), gNB((NNB + 63) / 64);

  // init: mess_bond = hardswish(init_bond @ w_bond + b)
  ga = {}; setP0(ib_bf, KP_BOND, KT160, nullptr, pk_wbond);
  ga.M = NBOND; ga.bias = b_bond; ga.out = mb_f; ga.outBf = mb_bf;
  wmma_gemm<1, EPI_HS_BF><<<gE, 256, 0, stream>>>(ga);

  // init: mess_node = hardswish(node @ w_node + b)
  ga = {}; setP0(node_bf, KP_NODE, KT160, nullptr, pk_wnode);
  ga.M = NNODE; ga.bias = b_node; ga.out = mn_f; ga.outBf = mn_bf;
  wmma_gemm<1, EPI_HS_BF><<<gN, 256, 0, stream>>>(ga);

  for (int layer = 0; layer < 3; ++layer) {
    // s_ij = segment_sum(mess_bond[ki], ij)
    zero(s_f, (size_t)NBOND * DD);
    seg_sum<<<(NNB * 64 + 255) / 256, 256, 0, stream>>>(mb_f, ki_idx, ij_idx, s_f, NNB);
    cvt(s_f, s_bf, (size_t)NBOND * DD);

    // z = sigmoid(init_bond @ wz_lo + s @ wz_hi + b_z)
    ga = {}; setP0(ib_bf, KP_BOND, KT160, nullptr, pk_wz_lo);
    setP1(s_bf, DD, KT256, nullptr, pk_wz_hi);
    ga.M = NBOND; ga.bias = b_z; ga.out = z_f;
    wmma_gemm<2, EPI_SIG><<<gE, 256, 0, stream>>>(ga);

    // r_ij = segsum(sigmoid(init_bond[ij]@wr_lo + mess[ki]@wr_hi + b_r) * mess[ki], ij)
    zero(r_f, (size_t)NBOND * DD);
    ga = {}; setP0(ib_bf, KP_BOND, KT160, ij_idx, pk_wr_lo);
    setP1(mb_bf, DD, KT256, ki_idx, pk_wr_hi);
    ga.M = NNB; ga.bias = b_r;
    ga.messF = mb_f; ga.gIJ = ij_idx; ga.gKI = ki_idx; ga.rDst = r_f;
    wmma_gemm<2, EPI_RF><<<gNB, 256, 0, stream>>>(ga);
    cvt(r_f, r_bf, (size_t)NBOND * DD);

    // m = tanh(init_bond @ w_m + r @ u + b_m); mess_bond = (1-z)*s + z*m
    ga = {}; setP0(ib_bf, KP_BOND, KT160, nullptr, pk_wm);
    setP1(r_bf, DD, KT256, nullptr, pk_u);
    ga.M = NBOND; ga.bias = b_m; ga.out = mb_f; ga.outBf = mb_bf;
    ga.zBuf = z_f; ga.sBuf = s_f;
    wmma_gemm<2, EPI_GRU><<<gE, 256, 0, stream>>>(ga);

    // aggr_node = segment_sum(mess_bond, j_idx)
    zero(ag_f, (size_t)NNODE * DD);
    seg_sum<<<(NBOND * 64 + 255) / 256, 256, 0, stream>>>(mb_f, nullptr, j_idx, ag_f, NBOND);
    cvt(ag_f, ag_bf, (size_t)NNODE * DD);

    // mess_node = hardswish(node@w_n + mess_node@un_lo + aggr@un_hi + b_n)
    ga = {}; setP0(node_bf, KP_NODE, KT160, nullptr, pk_wn);
    setP1(mn_bf, DD, KT256, nullptr, pk_un_lo);
    setP2(ag_bf, DD, KT256, nullptr, pk_un_hi);
    ga.M = NNODE; ga.bias = b_n; ga.out = mn_f; ga.outBf = mn_bf;
    wmma_gemm<3, EPI_HS_BF><<<gN, 256, 0, stream>>>(ga);
  }

  // final bond: hardswish(init_bond@wbf_lo + mess_bond@wbf_hi + b) -> d_out
  ga = {}; setP0(ib_bf, KP_BOND, KT160, nullptr, pk_wbf_lo);
  setP1(mb_bf, DD, KT256, nullptr, pk_wbf_hi);
  ga.M = NBOND; ga.bias = b_bond_final; ga.out = outBond;
  wmma_gemm<2, EPI_HS><<<gE, 256, 0, stream>>>(ga);

  // final node: hardswish(node@wnf_lo + mess_node@wnf_hi + b) -> d_out
  ga = {}; setP0(node_bf, KP_NODE, KT160, nullptr, pk_wnf_lo);
  setP1(mn_bf, DD, KT256, nullptr, pk_wnf_hi);
  ga.M = NNODE; ga.bias = b_node_final; ga.out = outNode;
  wmma_gemm<2, EPI_HS><<<gN, 256, 0, stream>>>(ga);
}
